// TimelineRNN_69965017251845
// MI455X (gfx1250) — matrix operand order repaired and checked by reading, not compiled, not run on hardware
//
#include <hip/hip_runtime.h>
#include <hip/hip_bf16.h>
#include <math.h>

// ---- problem constants (match reference) ----
#define Bsz   512
#define Ttxt  64
#define Edim  50
#define Hdim  32
#define Nseq  32768
#define Ttl   32
#define DNf   49
#define FCIN  305     // 49 + 4*64
#define HID   32

typedef __attribute__((ext_vector_type(16))) _Float16 v16h;
typedef __attribute__((ext_vector_type(8)))  _Float16 v8h;
typedef __attribute__((ext_vector_type(8)))  float    v8f;

#define WAVES     8
#define HS_STRIDE 48   // 16x32 f16 h-tile, padded rows (96B, 16B aligned)
#define WAVE_LDS  (16*HS_STRIDE)

// Native CDNA5 tanh (TRANS unit, co-executes with XDL WMMA).
__device__ __forceinline__ float fast_tanh(float x) {
#if __has_builtin(__builtin_amdgcn_tanhf)
  return __builtin_amdgcn_tanhf(x);
#else
  float r;
  asm volatile("v_tanh_f32 %0, %1" : "=v"(r) : "v"(x));
  return r;
#endif
}

// A-fragment (16x32 f16): lane holds row M = lane&15; element j holds
//   K = kchunk*32 + 8*hi + j          for j in 0..7
//   K = kchunk*32 + 16 + 8*hi + (j-8) for j in 8..15
// i.e. two contiguous 8-element runs per lane -> two 16B LDS loads.
__device__ __forceinline__ v16h lds_load_a(const _Float16* tile, int stride,
                                           int row, int hi) {
  const _Float16* base = tile + row * stride + 8 * hi;
  v8h lo = *(const v8h*)(base);        // ds_load_b128
  v8h hp = *(const v8h*)(base + 16);   // ds_load_b128
  v16h a;
#pragma unroll
  for (int i = 0; i < 8; ++i) { a[i] = lo[i]; a[i + 8] = hp[i]; }
  return a;
}

// Raw x row data for one timestep, loaded straight from global memory as
// aligned float2 (row base is 8B aligned: offset = (..*50 + even)*4B).
// Runs per lane: K0 = 8*hi, K1 = 16+8*hi (chunk 0, always valid since < 32)
//                K2 = 32+8*hi (valid, <= 47), K3 = 48+8*hi (only K=48,49 valid)
struct XRaw { float2 q[13]; };

__device__ __forceinline__ XRaw load_x_raw(const float* __restrict__ rowp, int hi) {
  XRaw r;
  const float* p0 = rowp + 8 * hi;
  const float* p1 = rowp + 16 + 8 * hi;
  const float* p2 = rowp + 32 + 8 * hi;
#pragma unroll
  for (int j = 0; j < 4; ++j) r.q[j]     = *(const float2*)(p0 + 2 * j);
#pragma unroll
  for (int j = 0; j < 4; ++j) r.q[4 + j] = *(const float2*)(p1 + 2 * j);
#pragma unroll
  for (int j = 0; j < 4; ++j) r.q[8 + j] = *(const float2*)(p2 + 2 * j);
  if (hi == 0) {                      // K = 48,49 (last two valid features)
    r.q[12] = *(const float2*)(rowp + 48);
  } else {
    r.q[12].x = 0.0f; r.q[12].y = 0.0f;
  }
  return r;
}

// Convert raw float2 runs into the two f16 A-fragments (K 0-31 and K 32-63).
__device__ __forceinline__ void cvt_frags(const XRaw& r, v16h& a0, v16h& a1) {
#pragma unroll
  for (int j = 0; j < 4; ++j) {
    a0[2 * j]     = (_Float16)r.q[j].x;     a0[2 * j + 1] = (_Float16)r.q[j].y;
    a0[8 + 2 * j] = (_Float16)r.q[4 + j].x; a0[9 + 2 * j] = (_Float16)r.q[4 + j].y;
    a1[2 * j]     = (_Float16)r.q[8 + j].x; a1[2 * j + 1] = (_Float16)r.q[8 + j].y;
  }
  a1[8] = (_Float16)r.q[12].x;  a1[9] = (_Float16)r.q[12].y;
#pragma unroll
  for (int j = 10; j < 16; ++j) a1[j] = (_Float16)0.0f;   // K >= 50 pad
}

// B-fragment (32x16 f16, K x N): lane n = nbase + (lane&15), element j holds
// K = kbase + 8*hi + j (j<8), kbase + 16 + 8*hi + (j-8) (j>=8). Source W row-major [32 x ncols]; B[k][n] = W[n][k].
__device__ __forceinline__ v16h load_b_w(const float* __restrict__ W, int ncols,
                                         int nbase, int kbase, int kvalid,
                                         int lo16, int hi) {
  const int n = nbase + lo16;
  v16h b;
#pragma unroll
  for (int j = 0; j < 16; ++j) {
    const int k = kbase + 8 * hi + j + ((j >= 8) ? 8 : 0);
    const float w = (k < kvalid) ? W[n * ncols + k] : 0.0f;
    b[j] = (_Float16)w;
  }
  return b;
}

// Bidirectional Elman RNN (tanh), final hidden only.
// One wave32 per 16-sequence tile, waves fully independent (no block barriers):
// the only LDS is the wave-private hidden tile used for the C->A transpose,
// and DS ops within one wave complete in order.
__global__ void __launch_bounds__(32 * WAVES)
birnn_kernel(const float* __restrict__ x, int nseq, int slen,
             const float* __restrict__ wif, const float* __restrict__ whf,
             const float* __restrict__ bif, const float* __restrict__ bhf,
             const float* __restrict__ wib, const float* __restrict__ whb,
             const float* __restrict__ bib, const float* __restrict__ bhb,
             float* __restrict__ hout /* [nseq, 2H] = [fwd | bwd] */) {
  __shared__ _Float16 lds[WAVES * WAVE_LDS];
  const int lane = threadIdx.x & 31;
  const int wave = threadIdx.x >> 5;
  const int hi   = lane >> 4;
  const int lo16 = lane & 15;
  _Float16* hs = lds + wave * WAVE_LDS;   // 16 x 32 (padded 48) f16 h-tile
  const int seq0 = (blockIdx.x * WAVES + wave) * 16;

  // this lane's x row (sequence seq0 + lo16)
  const float* xrow = x + (size_t)(seq0 + lo16) * slen * Edim;

  for (int d = 0; d < 2; ++d) {
    const float* wi = d ? wib : wif;
    const float* wh = d ? whb : whf;
    const float* bi = d ? bib : bif;
    const float* bh = d ? bhb : bhf;

    // weight fragments live in VGPRs across the whole time loop
    v16h bx00 = load_b_w(wi, Edim,  0, 0,  Edim, lo16, hi);  // n 0-15,  k 0-31
    v16h bx01 = load_b_w(wi, Edim,  0, 32, Edim, lo16, hi);  // n 0-15,  k 32-63
    v16h bx10 = load_b_w(wi, Edim, 16, 0,  Edim, lo16, hi);  // n 16-31, k 0-31
    v16h bx11 = load_b_w(wi, Edim, 16, 32, Edim, lo16, hi);  // n 16-31, k 32-63
    v16h bh0  = load_b_w(wh, Hdim,  0, 0,  Hdim, lo16, hi);  // n 0-15,  k 0-31
    v16h bh1  = load_b_w(wh, Hdim, 16, 0,  Hdim, lo16, hi);  // n 16-31, k 0-31

    const float bias0 = bi[lo16]      + bh[lo16];       // col n = lo16
    const float bias1 = bi[16 + lo16] + bh[16 + lo16];  // col n = 16 + lo16

    // h0 = 0 (pad cols 32..47 also 0, never read as K)
    for (int idx = lane; idx < 16 * HS_STRIDE; idx += 32)
      hs[idx] = (_Float16)0.0f;
    asm volatile("" ::: "memory");

    const int tstep = d ? -1 : 1;
    int tm = d ? (slen - 1) : 0;

    // software pipeline: raw x for step t already in flight/registers
    XRaw raw = load_x_raw(xrow + tm * Edim, hi);

    for (int t = 0; t < slen; ++t) {
      v16h ax0, ax1;
      cvt_frags(raw, ax0, ax1);          // waits on step-t loads here

      const int tn = tm + tstep;
      if (t + 1 < slen) raw = load_x_raw(xrow + tn * Edim, hi);   // overlap
      if (t + 2 < slen)
        __builtin_prefetch(xrow + (tn + tstep) * Edim, 0, 1);     // global_prefetch_b8

      asm volatile("" ::: "memory");
      v16h ah = lds_load_a(hs, HS_STRIDE, lo16, hi);   // prev hidden, K 0-31

      v8f acc0, acc1;
#pragma unroll
      for (int v = 0; v < 8; ++v) { acc0[v] = bias0; acc1[v] = bias1; }

      acc0 = __builtin_amdgcn_wmma_f32_16x16x32_f16(false, ax0, false, bx00, (short)0, acc0, false, false);
      acc0 = __builtin_amdgcn_wmma_f32_16x16x32_f16(false, ax1, false, bx01, (short)0, acc0, false, false);
      acc0 = __builtin_amdgcn_wmma_f32_16x16x32_f16(false, ah,  false, bh0,  (short)0, acc0, false, false);
      acc1 = __builtin_amdgcn_wmma_f32_16x16x32_f16(false, ax0, false, bx10, (short)0, acc1, false, false);
      acc1 = __builtin_amdgcn_wmma_f32_16x16x32_f16(false, ax1, false, bx11, (short)0, acc1, false, false);
      acc1 = __builtin_amdgcn_wmma_f32_16x16x32_f16(false, ah,  false, bh1,  (short)0, acc1, false, false);

      asm volatile("" ::: "memory");
      // C fragment: VGPR v -> row M = v + 8*hi, col N = lo16 (acc0) / 16+lo16 (acc1)
#pragma unroll
      for (int v = 0; v < 8; ++v) {
        const int row = v + 8 * hi;
        hs[row * HS_STRIDE + lo16]      = (_Float16)fast_tanh(acc0[v]);
        hs[row * HS_STRIDE + 16 + lo16] = (_Float16)fast_tanh(acc1[v]);
      }
      asm volatile("" ::: "memory");
      tm = tn;
    }

    // write final hidden for this direction: hout[:, d*32 : d*32+32]
    for (int idx = lane; idx < 16 * Hdim; idx += 32) {
      const int r = idx >> 5, c = idx & 31;
      hout[(size_t)(seq0 + r) * (2 * Hdim) + d * Hdim + c] =
          (float)hs[r * HS_STRIDE + c];
    }
    asm volatile("" ::: "memory");
  }
}

// Ragged segment mean/min/max over h2 [Nseq, 64]. One block per segment, one
// thread per feature column. Prefix-sum of counts recomputed per block (B=512).
__global__ void __launch_bounds__(64)
segstat_kernel(const float* __restrict__ h2, const int* __restrict__ cnt,
               float* __restrict__ smean, float* __restrict__ smin,
               float* __restrict__ smax) {
  const int b = blockIdx.x;
  const int c = threadIdx.x;  // 0..63
  int start = 0;
  for (int i = 0; i < b; ++i) start += cnt[i];
  const int n = cnt[b];
  float s = 0.0f, lo = 3.4e38f, hi = -3.4e38f;
  for (int i = 0; i < n; ++i) {
    const float v = h2[(size_t)(start + i) * 64 + c];
    s += v; lo = fminf(lo, v); hi = fmaxf(hi, v);
  }
  const float inv = (n > 0) ? (1.0f / (float)n) : 0.0f;
  smean[b * 64 + c] = s * inv;
  smin[b * 64 + c]  = lo;
  smax[b * 64 + c]  = hi;
}

// FC head: x = [h_n(64) | normal(49) | mean(64) | min(64) | max(64)] (305)
// y1 = tanh(x @ fc1_w^T + b1) [32]; out = sigmoid(y1 @ fc2_w^T + b2) [1]
__global__ void __launch_bounds__(32)
head_kernel(const float* __restrict__ hn, const float* __restrict__ normal,
            const float* __restrict__ smean, const float* __restrict__ smin,
            const float* __restrict__ smax,
            const float* __restrict__ fc1w, const float* __restrict__ fc1b,
            const float* __restrict__ fc2w, const float* __restrict__ fc2b,
            float* __restrict__ out) {
  __shared__ float xbuf[FCIN];
  __shared__ float h1[HID];
  const int row = blockIdx.x;
  const int tid = threadIdx.x;
  for (int i = tid; i < FCIN; i += 32) {
    float v;
    if (i < 64)        v = hn[row * 64 + i];
    else if (i < 113)  v = normal[row * DNf + (i - 64)];
    else if (i < 177)  v = smean[row * 64 + (i - 113)];
    else if (i < 241)  v = smin[row * 64 + (i - 177)];
    else               v = smax[row * 64 + (i - 241)];
    xbuf[i] = v;
  }
  __syncthreads();
  float acc = fc1b[tid];
  for (int k = 0; k < FCIN; ++k) acc += fc1w[tid * FCIN + k] * xbuf[k];
  h1[tid] = fast_tanh(acc);
  __syncthreads();
  if (tid == 0) {
    float a = fc2b[0];
    for (int k = 0; k < HID; ++k) a += fc2w[k] * h1[k];
    out[row] = 1.0f / (1.0f + expf(-a));
  }
}

extern "C" void kernel_launch(void* const* d_in, const int* in_sizes, int n_in,
                              void* d_out, int out_size, void* d_ws, size_t ws_size,
                              hipStream_t stream) {
  // setup_inputs() dict order
  const float* normal = (const float*)d_in[0];
  const float* textf  = (const float*)d_in[1];
  const float* tlf    = (const float*)d_in[2];
  const int*   tle    = (const int*)d_in[3];
  const float* r1wif = (const float*)d_in[4];
  const float* r1whf = (const float*)d_in[5];
  const float* r1bif = (const float*)d_in[6];
  const float* r1bhf = (const float*)d_in[7];
  const float* r1wib = (const float*)d_in[8];
  const float* r1whb = (const float*)d_in[9];
  const float* r1bib = (const float*)d_in[10];
  const float* r1bhb = (const float*)d_in[11];
  const float* r2wif = (const float*)d_in[12];
  const float* r2whf = (const float*)d_in[13];
  const float* r2bif = (const float*)d_in[14];
  const float* r2bhf = (const float*)d_in[15];
  const float* r2wib = (const float*)d_in[16];
  const float* r2whb = (const float*)d_in[17];
  const float* r2bib = (const float*)d_in[18];
  const float* r2bhb = (const float*)d_in[19];
  const float* fc1w  = (const float*)d_in[20];
  const float* fc1b  = (const float*)d_in[21];
  const float* fc2w  = (const float*)d_in[22];
  const float* fc2b  = (const float*)d_in[23];

  // workspace layout (fp32): h2[N,64] | hn[B,64] | mean/min/max [B,64] each
  float* h2    = (float*)d_ws;
  float* hn    = h2 + (size_t)Nseq * 64;
  float* smean = hn + (size_t)Bsz * 64;
  float* smin  = smean + (size_t)Bsz * 64;
  float* smax  = smin + (size_t)Bsz * 64;

  // 1) timeline bi-RNN (dominant): 2048 wave-tiles -> 256 blocks x 8 waves
  birnn_kernel<<<Nseq / (16 * WAVES), 32 * WAVES, 0, stream>>>(
      tlf, Nseq, Ttl, r2wif, r2whf, r2bif, r2bhf, r2wib, r2whb, r2bib, r2bhb, h2);
  // 2) text bi-RNN: 32 wave-tiles -> 4 blocks
  birnn_kernel<<<Bsz / (16 * WAVES), 32 * WAVES, 0, stream>>>(
      textf, Bsz, Ttxt, r1wif, r1whf, r1bif, r1bhf, r1wib, r1whb, r1bib, r1bhb, hn);
  // 3) ragged segment stats
  segstat_kernel<<<Bsz, 64, 0, stream>>>(h2, tle, smean, smin, smax);
  // 4) FC head
  head_kernel<<<Bsz, 32, 0, stream>>>(hn, normal, smean, smin, smax,
                                      fc1w, fc1b, fc2w, fc2b, (float*)d_out);
}
